// minGRU_68358699483578
// MI455X (gfx1250) — matrix operand order repaired and checked by reading, not compiled, not run on hardware
//
#include <hip/hip_runtime.h>

// Problem constants (from reference)
#define DIMX      1024
#define DIMI      2048
#define NBATCH    4
#define SEQ       4096
#define MTOT      (NBATCH * SEQ)     // 16384
#define NCHUNK    32
#define CHLEN     (SEQ / NCHUNK)     // 128

typedef __attribute__((ext_vector_type(16))) __bf16 v16bf;
typedef __attribute__((ext_vector_type(2)))  __bf16 v2bf;
typedef __attribute__((ext_vector_type(8)))  float  v8f;

struct __attribute__((aligned(8))) U64 { unsigned int x, y; };

// Native RTNE fp32->bf16, packed pair -> one uint (backend: v_cvt_pk_bf16_f32)
static __device__ __forceinline__ unsigned int pack2(float lo, float hi) {
  union { v2bf v; unsigned int u; } c;
  c.v[0] = (__bf16)lo;
  c.v[1] = (__bf16)hi;
  return c.u;
}
static __device__ __forceinline__ unsigned short f2bf16(float f) {
  union { __bf16 b; unsigned short u; } c;
  c.b = (__bf16)f;
  return c.u;
}

// Staging: raw global load type + pack-to-bf16-pair conversion (deferred)
template <typename T> struct Stage;
template <> struct Stage<float> {
  typedef float4 R;
  static __device__ __forceinline__ R   ld(const float* p) { return *(const float4*)p; }
  static __device__ __forceinline__ U64 pk(R r) {
    U64 u; u.x = pack2(r.x, r.y); u.y = pack2(r.z, r.w); return u;
  }
};
template <> struct Stage<unsigned short> {   // already-bf16 input
  typedef U64 R;
  static __device__ __forceinline__ R   ld(const unsigned short* p) { return *(const U64*)p; }
  static __device__ __forceinline__ U64 pk(R r) { return r; }
};

// ---------------------------------------------------------------------------
// Tiled GEMM: C[M][N] = A[M][K] * B[N][K]^T   (both row-major, contiguous K)
// Block tile 128x128, 8 waves (2M x 4N), wave tile 64x32, K-step 32.
// Double-buffered LDS (1 barrier/iter); raw global loads issued before the
// WMMA block, converted to bf16 and stored to the other buffer after it.
// ---------------------------------------------------------------------------
template <typename TA>
__global__ __launch_bounds__(256)
void mingru_gemm_tn(const TA* __restrict__ A, const float* __restrict__ Bw,
                    float* __restrict__ C, int M, int N, int K) {
  // padded row stride: 40 bf16 = 20 uints = 80 bytes (16B aligned, conflict-free)
  __shared__ unsigned int As[2][128][20];
  __shared__ unsigned int Bs[2][128][20];

  const int tid  = threadIdx.x;
  const int lane = tid & 31;
  const int wave = tid >> 5;
  const int wm   = (wave >> 2) * 64;   // wave M offset in tile
  const int wn   = (wave & 3) * 32;    // wave N offset in tile
  const int hl   = lane >> 4;          // half-wave
  const int r15  = lane & 15;
  const int m0   = blockIdx.y * 128;
  const int n0   = blockIdx.x * 128;

  v8f acc[4][2];
#pragma unroll
  for (int i = 0; i < 4; ++i)
#pragma unroll
    for (int j = 0; j < 2; ++j)
#pragma unroll
      for (int q = 0; q < 8; ++q) acc[i][j][q] = 0.0f;

  const int lrow = tid >> 3;          // 0..31 (+32 per pass)
  const int lcol = (tid & 7) * 4;     // 0..28 step 4
  const int lu   = lcol >> 1;         // uint column in LDS

  const TA*    Ap = A  + (size_t)(m0 + lrow) * K + lcol;
  const float* Bp = Bw + (size_t)(n0 + lrow) * K + lcol;
  const size_t rstepA = (size_t)32 * K;   // +32 rows

  // ---- prologue: stage tile 0 into buffer 0 ----
#pragma unroll
  for (int p = 0; p < 4; ++p) {
    *(U64*)&As[0][lrow + p * 32][lu] = Stage<TA>::pk(Stage<TA>::ld(Ap + p * rstepA));
    *(U64*)&Bs[0][lrow + p * 32][lu] = Stage<float>::pk(Stage<float>::ld(Bp + p * rstepA));
  }

  int buf = 0;
  for (int kk = 0; kk < K; kk += 32) {
    __syncthreads();   // buffer[buf] ready; buffer[buf^1] fully consumed

    // ---- issue next tile's global loads early (raw, no conversion yet) ----
    typename Stage<TA>::R    sa[4];
    typename Stage<float>::R sb[4];
    const bool more = (kk + 32) < K;
    if (more) {
#pragma unroll
      for (int p = 0; p < 4; ++p) {
        sa[p] = Stage<TA>::ld(Ap + p * rstepA + kk + 32);
        sb[p] = Stage<float>::ld(Bp + p * rstepA + kk + 32);
      }
      if (kk + 64 < K) {   // hint one stage further (global_prefetch_b8)
        __builtin_prefetch(Ap + kk + 64, 0, 0);
        __builtin_prefetch(Bp + kk + 64, 0, 0);
      }
    }

    // ---- compute from buffer[buf] ----
    union Frag { v16bf v; U64 d[4]; };
    Frag afr[4], bfr[2];
    const int ak = hl * 4;  // A: K-base = hl*8 bf16  (lanes 0-15: K0-7/16-23)
#pragma unroll
    for (int i = 0; i < 4; ++i) {
      const int row = wm + i * 16 + r15;
      afr[i].d[0] = *(const U64*)&As[buf][row][ak];
      afr[i].d[1] = *(const U64*)&As[buf][row][ak + 2];
      afr[i].d[2] = *(const U64*)&As[buf][row][ak + 8];   // K base + 16
      afr[i].d[3] = *(const U64*)&As[buf][row][ak + 10];
    }
    const int bk = hl * 4;  // B: K-base = hl*8 bf16 (lanes 0-15: K0-7/16-23, lane = column)
#pragma unroll
    for (int j = 0; j < 2; ++j) {
      const int col = wn + j * 16 + r15;
      bfr[j].d[0] = *(const U64*)&Bs[buf][col][bk];
      bfr[j].d[1] = *(const U64*)&Bs[buf][col][bk + 2];
      bfr[j].d[2] = *(const U64*)&Bs[buf][col][bk + 8];
      bfr[j].d[3] = *(const U64*)&Bs[buf][col][bk + 10];
    }
#pragma unroll
    for (int i = 0; i < 4; ++i)
#pragma unroll
      for (int j = 0; j < 2; ++j)
        acc[i][j] = __builtin_amdgcn_wmma_f32_16x16x32_bf16(
            false, afr[i].v, false, bfr[j].v, (short)0, acc[i][j], false, false);

    // ---- convert + store staged tile into the other buffer ----
    if (more) {
#pragma unroll
      for (int p = 0; p < 4; ++p) {
        *(U64*)&As[buf ^ 1][lrow + p * 32][lu] = Stage<TA>::pk(sa[p]);
        *(U64*)&Bs[buf ^ 1][lrow + p * 32][lu] = Stage<float>::pk(sb[p]);
      }
    }
    buf ^= 1;
  }

  // ---- writeback: D layout (VGPR r -> M = r + hl*8, lane -> N) ----
#pragma unroll
  for (int i = 0; i < 4; ++i) {
    const int mrow = m0 + wm + i * 16 + hl * 8;
#pragma unroll
    for (int j = 0; j < 2; ++j) {
      const int ncol = n0 + wn + j * 16 + r15;
#pragma unroll
      for (int r = 0; r < 8; ++r)
        C[(size_t)(mrow + r) * N + ncol] = acc[i][j][r];
    }
  }
}

// ---------------------------------------------------------------------------
// Elementwise: hg -> (log_coeffs, log_values), in place
// ---------------------------------------------------------------------------
static __device__ __forceinline__ float softplusf(float z) {
  return fmaxf(z, 0.0f) + log1pf(__expf(-fabsf(z)));
}
static __device__ __forceinline__ float laddexp(float a, float b) {
  float mx = fmaxf(a, b);
  if (mx == -__builtin_inff()) return mx;
  return mx + log1pf(__expf(fminf(a, b) - mx));
}

__global__ __launch_bounds__(256)
void mingru_elementwise(float* __restrict__ hg) {
  size_t gid = (size_t)blockIdx.x * 256 + threadIdx.x;
  int    e   = (int)(gid % DIMI);
  size_t m   = gid / DIMI;
  size_t bh  = m * (size_t)(2 * DIMI);
  float hidden = hg[bh + e];
  float gate   = hg[bh + DIMI + e];
  float lc = -softplusf(gate);                                  // log_coeffs
  float lz = -softplusf(-gate);                                 // log_z
  float lg = (hidden >= 0.0f) ? __logf(hidden + 0.5f) : -softplusf(-hidden);
  hg[bh + e]        = lc;
  hg[bh + DIMI + e] = lz + lg;                                  // log_values
}

// ---------------------------------------------------------------------------
// Chunked parallel scan: s_t = logaddexp(lc_t + s_{t-1}, lv_t), s_{-1} = -inf
// Chunk map: s_out = logaddexp(Csum + s_in, Vend)
// ---------------------------------------------------------------------------
__global__ __launch_bounds__(256)
void mingru_scan_chunks(const float* __restrict__ hg,
                        float* __restrict__ Cc, float* __restrict__ Vc) {
  size_t gid = (size_t)blockIdx.x * 256 + threadIdx.x;
  int e    = (int)(gid % DIMI);
  int rest = (int)(gid / DIMI);
  int ch   = rest % NCHUNK;
  int b    = rest / NCHUNK;
  size_t base = ((size_t)b * SEQ + (size_t)ch * CHLEN) * (size_t)(2 * DIMI);
  float cs = 0.0f, v = -__builtin_inff();
  for (int t = 0; t < CHLEN; ++t) {
    float lc = hg[base + (size_t)t * (2 * DIMI) + e];
    float lv = hg[base + (size_t)t * (2 * DIMI) + DIMI + e];
    cs += lc;
    v = laddexp(lc + v, lv);
  }
  size_t idx = ((size_t)b * NCHUNK + ch) * DIMI + e;
  Cc[idx] = cs;
  Vc[idx] = v;
}

__global__ __launch_bounds__(256)
void mingru_scan_prefix(const float* __restrict__ Cc, const float* __restrict__ Vc,
                        float* __restrict__ Sin) {
  size_t gid = (size_t)blockIdx.x * 256 + threadIdx.x;
  int e = (int)(gid % DIMI);
  int b = (int)(gid / DIMI);
  float s = -__builtin_inff();
  for (int ch = 0; ch < NCHUNK; ++ch) {
    size_t idx = ((size_t)b * NCHUNK + ch) * DIMI + e;
    Sin[idx] = s;
    s = laddexp(Cc[idx] + s, Vc[idx]);
  }
}

__global__ __launch_bounds__(256)
void mingru_scan_apply(const float* __restrict__ hg, const float* __restrict__ Sin,
                       unsigned short* __restrict__ H) {
  size_t gid = (size_t)blockIdx.x * 256 + threadIdx.x;
  int e    = (int)(gid % DIMI);
  int rest = (int)(gid / DIMI);
  int ch   = rest % NCHUNK;
  int b    = rest / NCHUNK;
  size_t base = ((size_t)b * SEQ + (size_t)ch * CHLEN) * (size_t)(2 * DIMI);
  float s = Sin[((size_t)b * NCHUNK + ch) * DIMI + e];
  size_t ho = ((size_t)b * SEQ + (size_t)ch * CHLEN) * DIMI + e;
  for (int t = 0; t < CHLEN; ++t) {
    float lc = hg[base + (size_t)t * (2 * DIMI) + e];
    float lv = hg[base + (size_t)t * (2 * DIMI) + DIMI + e];
    s = laddexp(lc + s, lv);
    H[ho + (size_t)t * DIMI] = f2bf16(__expf(s));   // h = exp(log_h)
  }
}

// ---------------------------------------------------------------------------
extern "C" void kernel_launch(void* const* d_in, const int* in_sizes, int n_in,
                              void* d_out, int out_size, void* d_ws, size_t ws_size,
                              hipStream_t stream) {
  const float* x    = (const float*)d_in[0];   // [4,4096,1024]
  const float* Whg  = (const float*)d_in[1];   // [4096,1024]
  const float* Wout = (const float*)d_in[2];   // [1024,2048]
  float*       out  = (float*)d_out;           // [4,4096,1024]

  char* ws = (char*)d_ws;
  float* hg = (float*)ws;                                     // 16384 x 4096 f32 (256 MB)
  size_t off = (size_t)MTOT * 2 * DIMI * sizeof(float);
  float* Cc  = (float*)(ws + off); off += (size_t)NBATCH * NCHUNK * DIMI * sizeof(float);
  float* Vc  = (float*)(ws + off); off += (size_t)NBATCH * NCHUNK * DIMI * sizeof(float);
  float* Sin = (float*)(ws + off); off += (size_t)NBATCH * NCHUNK * DIMI * sizeof(float);
  unsigned short* H = (unsigned short*)(ws + off);            // 16384 x 2048 bf16 (64 MB)

  dim3 blk(256);

  // GEMM1: hg = x @ W_hg^T   (M=16384, N=4096, K=1024)
  mingru_gemm_tn<float><<<dim3((2 * DIMI) / 128, MTOT / 128), blk, 0, stream>>>(
      x, Whg, hg, MTOT, 2 * DIMI, DIMX);

  // log_coeffs / log_values, in place
  mingru_elementwise<<<(MTOT * (size_t)DIMI) / 256, blk, 0, stream>>>(hg);

  // chunked scan over sequence
  mingru_scan_chunks<<<((size_t)NBATCH * NCHUNK * DIMI) / 256, blk, 0, stream>>>(hg, Cc, Vc);
  mingru_scan_prefix<<<((size_t)NBATCH * DIMI) / 256, blk, 0, stream>>>(Cc, Vc, Sin);
  mingru_scan_apply<<<((size_t)NBATCH * NCHUNK * DIMI) / 256, blk, 0, stream>>>(hg, Sin, H);

  // GEMM2: out = h @ W_out^T  (M=16384, N=1024, K=2048), A already bf16
  mingru_gemm_tn<unsigned short><<<dim3(DIMX / 128, MTOT / 128), blk, 0, stream>>>(
      H, Wout, out, MTOT, DIMX, DIMI);
}
